// UniTransformerO2TwoUpdateGeneral_53412213293451
// MI455X (gfx1250) — matrix operand order repaired and checked by reading, not compiled, not run on hardware
//
#include <hip/hip_runtime.h>
#include <hip/hip_bf16.h>

// Fused UniTransformerO2TwoUpdateGeneral for MI455X (gfx1250, wave32, WMMA).
// All heavy GEMMs (per-edge and per-node 2-layer MLPs) run on
// v_wmma_f32_16x16x32_f16; LayerNorm is done in-wave with shuffles.

typedef _Float16 h16;
typedef __attribute__((ext_vector_type(16))) _Float16 v16h;
typedef __attribute__((ext_vector_type(8)))  _Float16 v8h;
typedef __attribute__((ext_vector_type(8)))  float    v8f;

#define N_NODES 8192
#define K_NN 24
#define E_EDGES (N_NODES * K_NN)
#define HID 128
#define HEADS 16
#define HEAD_D 8
#define NRG 20
#define EA_D 5
#define RFEAT 100
#define KV_IN 356
#define KV_PAD 384

#define DEV static __device__ __forceinline__

// ---------------- WMMA fragment loaders (per CDNA5 ISA 7.12.2 layouts) -----

// A: 16x32 f16, row-major tile in LDS. lanes 0-15: M=lane, K pairs at
// k0+{0..7} (VGPR0-3) and k0+16+{0..7} (VGPR4-7); lanes 16-31: +8 on K base.
DEV v16h frag_a_lds(const h16* __restrict__ tile, int ld, int m0, int k0) {
  const int l = threadIdx.x & 31;
  const h16* p = tile + (m0 + (l & 15)) * ld + k0 + ((l >> 4) << 3);
  const v8h lo = *(const v8h*)p;
  const v8h hi = *(const v8h*)(p + 16);
  v16h r;
#pragma unroll
  for (int i = 0; i < 8; ++i) { r[i] = lo[i]; r[8 + i] = hi[i]; }
  return r;
}

// B: 32x16 f16 from pre-transposed weights wt[N][Kpad] (row-major) in global.
// lanes 0-15 hold K=k0+{0..7},k0+16+{0..7} at column n; lanes 16-31: +8 on K base.
DEV v16h frag_b_glb(const h16* __restrict__ wt, int ldk, int n0, int k0) {
  const int l = threadIdx.x & 31;
  const h16* p = wt + (long)(n0 + (l & 15)) * ldk + k0 + ((l >> 4) << 3);
  const v8h lo = *(const v8h*)p;
  const v8h hi = *(const v8h*)(p + 16);
  v16h r;
#pragma unroll
  for (int i = 0; i < 8; ++i) { r[i] = lo[i]; r[8 + i] = hi[i]; }
  return r;
}

// ---------------- wave-level fused 2-layer MLP (Linear-LN-ReLU-Linear) -----
// One wave computes 16 rows x all columns. C layout: lane l, acc a, reg v:
// row = m0 + v + 8*(l>=16), col = a*16 + (l&15). Rows live entirely inside a
// 16-lane half -> LayerNorm reduces with shfl_xor 1,2,4,8.
// MODE 0: f16 store (optional per-row escale). MODE 1: f32 store + residual.
template <int K1, int N2, int MODE>
DEV void mlp2_wave(const h16* __restrict__ A, int ldA, int m0,
                   const h16* __restrict__ w1t, const float* __restrict__ b1,
                   const float* __restrict__ g1, const float* __restrict__ be1,
                   const h16* __restrict__ w2t, const float* __restrict__ b2,
                   h16* __restrict__ h1, h16* __restrict__ out16,
                   float* __restrict__ out32, long row0,
                   const float* escale, const float* __restrict__ resid) {
  const int l = threadIdx.x & 31;
  const int ncol = l & 15;
  const int hi8 = (l >> 4) << 3;

  v8f acc[8];
#pragma unroll
  for (int a = 0; a < 8; ++a)
#pragma unroll
    for (int i = 0; i < 8; ++i) acc[a][i] = 0.f;

#pragma unroll
  for (int kk = 0; kk < K1; kk += 32) {
    const v16h af = frag_a_lds(A, ldA, m0, kk);
#pragma unroll
    for (int a = 0; a < 8; ++a) {
      const v16h bf = frag_b_glb(w1t, K1, a * 16, kk);
      acc[a] = __builtin_amdgcn_wmma_f32_16x16x32_f16(false, af, false, bf,
                                                      (short)0, acc[a], false, false);
    }
  }
  // bias1
#pragma unroll
  for (int a = 0; a < 8; ++a) {
    const float bb = b1[a * 16 + ncol];
#pragma unroll
    for (int v = 0; v < 8; ++v) acc[a][v] += bb;
  }
  // LayerNorm row stats (row split across 16 lanes x 8 accs)
  float mu[8], rs[8];
#pragma unroll
  for (int v = 0; v < 8; ++v) {
    float s = 0.f, q = 0.f;
#pragma unroll
    for (int a = 0; a < 8; ++a) { const float x = acc[a][v]; s += x; q += x * x; }
#pragma unroll
    for (int m = 1; m <= 8; m <<= 1) {
      s += __shfl_xor(s, m, 32);
      q += __shfl_xor(q, m, 32);
    }
    const float mean = s * (1.f / 128.f);
    mu[v] = mean;
    rs[v] = __frsqrt_rn(q * (1.f / 128.f) - mean * mean + 1e-5f);
  }
  // normalize + affine + ReLU -> wave-private h1 rows (f16 in LDS)
#pragma unroll
  for (int a = 0; a < 8; ++a) {
    const int n = a * 16 + ncol;
    const float gg = g1[n], bb = be1[n];
#pragma unroll
    for (int v = 0; v < 8; ++v) {
      float x = (acc[a][v] - mu[v]) * rs[v] * gg + bb;
      x = fmaxf(x, 0.f);
      h1[(m0 + v + hi8) * HID + n] = (h16)x;
    }
  }
  // GEMM2: 16 x N2, K = 128
  constexpr int NA2 = N2 / 16;
  v8f acc2[NA2];
#pragma unroll
  for (int a = 0; a < NA2; ++a)
#pragma unroll
    for (int i = 0; i < 8; ++i) acc2[a][i] = 0.f;
#pragma unroll
  for (int kk = 0; kk < HID; kk += 32) {
    const v16h af = frag_a_lds(h1, HID, m0, kk);
#pragma unroll
    for (int a = 0; a < NA2; ++a) {
      const v16h bf = frag_b_glb(w2t, HID, a * 16, kk);
      acc2[a] = __builtin_amdgcn_wmma_f32_16x16x32_f16(false, af, false, bf,
                                                       (short)0, acc2[a], false, false);
    }
  }
#pragma unroll
  for (int a = 0; a < NA2; ++a) {
    const float bb = b2[a * 16 + ncol];
#pragma unroll
    for (int v = 0; v < 8; ++v) {
      const int r = m0 + v + hi8;
      const long gr = row0 + r;
      float x = acc2[a][v] + bb;
      if (MODE == 0) {
        if (escale) x *= escale[r];
        out16[gr * (long)N2 + a * 16 + ncol] = (h16)x;
      } else {
        const long idx = gr * (long)HID + a * 16 + ncol;
        out32[idx] = x + resid[idx];
      }
    }
  }
}

// ---------------- weight convert: w[K][N] f32 -> wt[N][Kpad] f16 -----------
__global__ void wconv_kernel(const float* __restrict__ w, h16* __restrict__ wt,
                             int K, int Nn, int Kpad) {
  const int i = blockIdx.x * 256 + threadIdx.x;
  if (i >= Nn * Kpad) return;
  const int n = i / Kpad, k = i % Kpad;
  wt[i] = (h16)((k < K) ? w[(long)k * Nn + n] : 0.f);
}

// ---------------- edge kernel: RBF + gathers + k/v MLPs (WMMA) -------------
template <int VN2>
__global__ __launch_bounds__(128) void edge_mlp_kernel(
    const float* __restrict__ xw, const float* __restrict__ hw,
    const float* __restrict__ edge_attr, const int* __restrict__ ei,
    const float* __restrict__ ew_w, const float* __restrict__ ew_b,
    const h16* __restrict__ kw1t, const float* __restrict__ kb1,
    const float* __restrict__ kg, const float* __restrict__ kbe,
    const h16* __restrict__ kw2t, const float* __restrict__ kb2,
    const h16* __restrict__ vw1t, const float* __restrict__ vb1,
    const float* __restrict__ vg, const float* __restrict__ vbe,
    const h16* __restrict__ vw2t, const float* __restrict__ vb2,
    h16* __restrict__ k_out, h16* __restrict__ v_out) {
  extern __shared__ __align__(16) char smem[];
  h16* kv = (h16*)smem;                       // [64][KV_PAD]
  h16* h1 = kv + 64 * KV_PAD;                 // [64][HID]
  float* ew_s = (float*)(h1 + 64 * HID);      // [64]
  const int t = threadIdx.x;
  const long e0 = (long)blockIdx.x * 64;

  if (t < 64) {
    const long e = e0 + t;
    const int s = ei[e];
    const int d = ei[E_EDGES + e];
    const float rx = xw[d * 3 + 0] - xw[s * 3 + 0];
    const float ry = xw[d * 3 + 1] - xw[s * 3 + 1];
    const float rz = xw[d * 3 + 2] - xw[s * 3 + 2];
    const float dist = sqrtf(rx * rx + ry * ry + rz * rz);
    float ea[EA_D];
#pragma unroll
    for (int c = 0; c < EA_D; ++c) ea[c] = edge_attr[e * EA_D + c];
    const float step = 10.f / 19.f;
    const float coeff = -0.5f / (step * step);
    float ew = ew_b[0];
#pragma unroll 1
    for (int a = 0; a < EA_D; ++a) {
#pragma unroll
      for (int gI = 0; gI < NRG; ++gI) {
        const float dd = dist - (float)gI * step;
        const float val = ea[a] * __expf(coeff * dd * dd);
        const int c = a * NRG + gI;
        kv[t * KV_PAD + c] = (h16)val;
        ew += val * ew_w[c];
      }
    }
    ew_s[t] = 1.f / (1.f + __expf(-ew));
#pragma unroll
    for (int c = KV_IN; c < KV_PAD; ++c) kv[t * KV_PAD + c] = (h16)0.f;
  }
  {
    const int r = t >> 1, part = t & 1;
    const long e = e0 + r;
    const int s = ei[e];
    const int d = ei[E_EDGES + e];
    const float4* hd = (const float4*)(hw + (long)d * HID);
    const float4* hs = (const float4*)(hw + (long)s * HID);
#pragma unroll
    for (int j = 0; j < 16; ++j) {
      const float4 f = hd[part * 16 + j];
      h16* p = kv + r * KV_PAD + RFEAT + part * 64 + j * 4;
      p[0] = (h16)f.x; p[1] = (h16)f.y; p[2] = (h16)f.z; p[3] = (h16)f.w;
    }
#pragma unroll
    for (int j = 0; j < 16; ++j) {
      const float4 f = hs[part * 16 + j];
      h16* p = kv + r * KV_PAD + RFEAT + HID + part * 64 + j * 4;
      p[0] = (h16)f.x; p[1] = (h16)f.y; p[2] = (h16)f.z; p[3] = (h16)f.w;
    }
  }
  __syncthreads();
  const int m0 = (t >> 5) * 16;
  mlp2_wave<KV_PAD, HID, 0>(kv, KV_PAD, m0, kw1t, kb1, kg, kbe, kw2t, kb2, h1,
                            k_out, nullptr, e0, nullptr, nullptr);
  mlp2_wave<KV_PAD, VN2, 0>(kv, KV_PAD, m0, vw1t, vb1, vg, vbe, vw2t, vb2, h1,
                            v_out, nullptr, e0, ew_s, nullptr);
}

// ---------------- node q MLP (WMMA) ----------------------------------------
__global__ __launch_bounds__(128) void node_mlp_kernel(
    const float* __restrict__ hw, const h16* __restrict__ w1t,
    const float* __restrict__ b1, const float* __restrict__ g,
    const float* __restrict__ be, const h16* __restrict__ w2t,
    const float* __restrict__ b2, h16* __restrict__ out) {
  __shared__ h16 kv[64 * HID];
  __shared__ h16 h1[64 * HID];
  const int t = threadIdx.x;
  const long n0 = (long)blockIdx.x * 64;
  {
    const int r = t >> 1, part = t & 1;
    const float4* hr = (const float4*)(hw + (n0 + r) * HID);
#pragma unroll
    for (int j = 0; j < 16; ++j) {
      const float4 f = hr[part * 16 + j];
      h16* p = kv + r * HID + part * 64 + j * 4;
      p[0] = (h16)f.x; p[1] = (h16)f.y; p[2] = (h16)f.z; p[3] = (h16)f.w;
    }
  }
  __syncthreads();
  mlp2_wave<HID, HID, 0>(kv, HID, (t >> 5) * 16, w1t, b1, g, be, w2t, b2, h1,
                         out, nullptr, n0, nullptr, nullptr);
}

// ---------------- out MLP: concat(agg,h) -> h (in-place residual) ----------
__global__ __launch_bounds__(128) void out_mlp_kernel(
    const float* __restrict__ agg, float* __restrict__ hw,
    const h16* __restrict__ w1t, const float* __restrict__ b1,
    const float* __restrict__ g, const float* __restrict__ be,
    const h16* __restrict__ w2t, const float* __restrict__ b2) {
  __shared__ h16 kv[64 * 256];
  __shared__ h16 h1[64 * HID];
  const int t = threadIdx.x;
  const long n0 = (long)blockIdx.x * 64;
  {
    const int r = t >> 1, part = t & 1;
    const float4* srcp = part ? (const float4*)(hw + (n0 + r) * HID)
                              : (const float4*)(agg + (n0 + r) * HID);
#pragma unroll
    for (int j = 0; j < 32; ++j) {
      const float4 f = srcp[j];
      h16* p = kv + r * 256 + part * 128 + j * 4;
      p[0] = (h16)f.x; p[1] = (h16)f.y; p[2] = (h16)f.z; p[3] = (h16)f.w;
    }
  }
  __syncthreads();
  mlp2_wave<256, HID, 1>(kv, 256, (t >> 5) * 16, w1t, b1, g, be, w2t, b2, h1,
                         nullptr, hw, n0, nullptr, hw);
}

// ---------------- attention (dst-contiguous: 24 edges per node) ------------
__global__ __launch_bounds__(256) void attn_x2h_kernel(
    const h16* __restrict__ q, const h16* __restrict__ k,
    const h16* __restrict__ v, float* __restrict__ agg) {
  const int tid = blockIdx.x * 256 + threadIdx.x;
  const int node = tid >> 4, hh = tid & 15;
  float qf[HEAD_D];
  {
    const v8h qv = *(const v8h*)(q + (long)node * HID + hh * 8);
#pragma unroll
    for (int d = 0; d < HEAD_D; ++d) qf[d] = (float)qv[d];
  }
  const float sc = 0.35355339059327373f;  // 1/sqrt(8)
  float lg[K_NN];
  float mx = -1e30f;
#pragma unroll 1
  for (int e = 0; e < K_NN; ++e) {
    const v8h kr = *(const v8h*)(k + ((long)node * K_NN + e) * HID + hh * 8);
    float s = 0.f;
#pragma unroll
    for (int d = 0; d < HEAD_D; ++d) s += qf[d] * (float)kr[d];
    s *= sc;
    lg[e] = s;
    mx = fmaxf(mx, s);
  }
  float ssum = 0.f;
#pragma unroll
  for (int e = 0; e < K_NN; ++e) { lg[e] = __expf(lg[e] - mx); ssum += lg[e]; }
  const float inv = 1.f / (ssum + 1e-16f);
  float av[HEAD_D];
#pragma unroll
  for (int d = 0; d < HEAD_D; ++d) av[d] = 0.f;
#pragma unroll 1
  for (int e = 0; e < K_NN; ++e) {
    const float al = lg[e] * inv;
    const v8h vr = *(const v8h*)(v + ((long)node * K_NN + e) * HID + hh * 8);
#pragma unroll
    for (int d = 0; d < HEAD_D; ++d) av[d] += al * (float)vr[d];
  }
#pragma unroll
  for (int d = 0; d < HEAD_D; ++d) agg[(long)node * HID + hh * 8 + d] = av[d];
}

__global__ __launch_bounds__(256) void attn_h2x_kernel(
    const h16* __restrict__ q, const h16* __restrict__ k,
    const h16* __restrict__ v /*[E][16]*/, const float* __restrict__ xw,
    const int* __restrict__ ei, float* __restrict__ dxh) {
  const int tid = blockIdx.x * 256 + threadIdx.x;
  const int node = tid >> 4, hh = tid & 15;
  float qf[HEAD_D];
  {
    const v8h qv = *(const v8h*)(q + (long)node * HID + hh * 8);
#pragma unroll
    for (int d = 0; d < HEAD_D; ++d) qf[d] = (float)qv[d];
  }
  const float sc = 0.35355339059327373f;
  float lg[K_NN];
  float mx = -1e30f;
#pragma unroll 1
  for (int e = 0; e < K_NN; ++e) {
    const v8h kr = *(const v8h*)(k + ((long)node * K_NN + e) * HID + hh * 8);
    float s = 0.f;
#pragma unroll
    for (int d = 0; d < HEAD_D; ++d) s += qf[d] * (float)kr[d];
    s *= sc;
    lg[e] = s;
    mx = fmaxf(mx, s);
  }
  float ssum = 0.f;
#pragma unroll
  for (int e = 0; e < K_NN; ++e) { lg[e] = __expf(lg[e] - mx); ssum += lg[e]; }
  const float inv = 1.f / (ssum + 1e-16f);
  const float xd0 = xw[node * 3 + 0], xd1 = xw[node * 3 + 1], xd2 = xw[node * 3 + 2];
  float dx0 = 0.f, dx1 = 0.f, dx2 = 0.f;
#pragma unroll 1
  for (int e = 0; e < K_NN; ++e) {
    const long ge = (long)node * K_NN + e;
    const float w = lg[e] * inv * (float)v[ge * HEADS + hh];
    const int s = ei[ge];
    dx0 += w * (xd0 - xw[s * 3 + 0]);
    dx1 += w * (xd1 - xw[s * 3 + 1]);
    dx2 += w * (xd2 - xw[s * 3 + 2]);
  }
  const long o = ((long)node * HEADS + hh) * 3;
  dxh[o + 0] = dx0; dxh[o + 1] = dx1; dxh[o + 2] = dx2;
}

__global__ void x_update_kernel(float* __restrict__ xw, const float* __restrict__ dxh) {
  const int node = blockIdx.x * 256 + threadIdx.x;
  if (node >= N_NODES) return;
#pragma unroll
  for (int c = 0; c < 3; ++c) {
    float s = 0.f;
    for (int hh = 0; hh < HEADS; ++hh) s += dxh[((long)node * HEADS + hh) * 3 + c];
    xw[node * 3 + c] += s * (1.f / HEADS);
  }
}

__global__ void init_copy_kernel(const float* __restrict__ hi, const float* __restrict__ xi,
                                 float* __restrict__ hb, float* __restrict__ xb) {
  const long i = (long)blockIdx.x * 256 + threadIdx.x;
  if (i < (long)N_NODES * HID) hb[i] = hi[i];
  if (i < (long)N_NODES * 3) xb[i] = xi[i];
}

__global__ void final_copy_kernel(const float* __restrict__ hb, const float* __restrict__ xb,
                                  float* __restrict__ o) {
  const long i = (long)blockIdx.x * 256 + threadIdx.x;
  if (i < (long)N_NODES * HID) o[i] = hb[i];
  if (i < (long)N_NODES * 3) o[(long)N_NODES * HID + i] = xb[i];
}

// ---------------------------------------------------------------------------
struct MlpDev { const float *b1, *b2, *be, *g; const h16 *w1t, *w2t; };
struct LayerDev {
  const float *x2h_eww, *x2h_ewb, *h2x_eww, *h2x_ewb;
  MlpDev hk, hq, hv, outm, xk, xq, xv;
};

extern "C" void kernel_launch(void* const* d_in, const int* in_sizes, int n_in,
                              void* d_out, int out_size, void* d_ws, size_t ws_size,
                              hipStream_t stream) {
  (void)in_sizes; (void)n_in; (void)out_size; (void)ws_size;
  const float* h_in = (const float*)d_in[0];
  const float* x_in = (const float*)d_in[1];
  const float* edge_attr = (const float*)d_in[2];
  const int* ei = (const int*)d_in[3];  // [2][E]: src then dst

  // workspace carve
  char* ws = (char*)d_ws;
  size_t off = 0;
  auto alloc = [&](size_t bytes) -> char* {
    char* p = ws + off;
    off = (off + bytes + 255) & ~(size_t)255;
    return p;
  };
  float* hb = (float*)alloc((size_t)N_NODES * HID * 4);
  float* xb = (float*)alloc((size_t)N_NODES * 3 * 4);
  h16* qb = (h16*)alloc((size_t)N_NODES * HID * 2);
  h16* kb = (h16*)alloc((size_t)E_EDGES * HID * 2);
  h16* vb = (h16*)alloc((size_t)E_EDGES * HID * 2);
  float* aggb = (float*)alloc((size_t)N_NODES * HID * 4);
  float* dxh = (float*)alloc((size_t)N_NODES * HEADS * 3 * 4);

  auto F = [&](int i) { return (const float*)d_in[i]; };
  auto conv = [&](const float* w, int K, int Nn, int Kpad) -> const h16* {
    h16* wt = (h16*)alloc((size_t)Nn * Kpad * 2);
    const int total = Nn * Kpad;
    wconv_kernel<<<(total + 255) / 256, 256, 0, stream>>>(w, wt, K, Nn, Kpad);
    return wt;
  };
  // MLP leaf order (sorted dict keys): b1, b2, be, g, w1, w2
  auto mk = [&](int idx, int K, int Kpad, int N2) -> MlpDev {
    MlpDev m;
    m.b1 = F(idx + 0); m.b2 = F(idx + 1); m.be = F(idx + 2); m.g = F(idx + 3);
    m.w1t = conv(F(idx + 4), K, HID, Kpad);
    m.w2t = conv(F(idx + 5), HID, N2, HID);
    return m;
  };

  // params pytree order: layers[L] -> {h2x, x2h} sorted; h2x: ew_b, ew_w,
  // xk, xq, xv; x2h: ew_b, ew_w, hk, hq, hv, out. 46 leaves per layer.
  LayerDev L[2];
  for (int i = 0; i < 2; ++i) {
    const int b = 5 + 46 * i;
    L[i].h2x_ewb = F(b + 0);  L[i].h2x_eww = F(b + 1);
    L[i].xk = mk(b + 2, KV_IN, KV_PAD, HID);
    L[i].xq = mk(b + 8, HID, HID, HID);
    L[i].xv = mk(b + 14, KV_IN, KV_PAD, HEADS);
    L[i].x2h_ewb = F(b + 20); L[i].x2h_eww = F(b + 21);
    L[i].hk = mk(b + 22, KV_IN, KV_PAD, HID);
    L[i].hq = mk(b + 28, HID, HID, HID);
    L[i].hv = mk(b + 34, KV_IN, KV_PAD, HID);
    L[i].outm = mk(b + 40, 2 * HID, 2 * HID, HID);
  }

  const int copy_blocks = (N_NODES * HID + 255) / 256;
  init_copy_kernel<<<copy_blocks, 256, 0, stream>>>(h_in, x_in, hb, xb);

  const size_t esm = (size_t)64 * KV_PAD * 2 + (size_t)64 * HID * 2 + 64 * 4;
  const int eblocks = E_EDGES / 64;
  const int nblocks = N_NODES / 64;
  const int ablocks = N_NODES * HEADS / 256;

  for (int i = 0; i < 2; ++i) {
    const LayerDev& P = L[i];
    // ---- x2h ----
    node_mlp_kernel<<<nblocks, 128, 0, stream>>>(hb, P.hq.w1t, P.hq.b1, P.hq.g,
                                                 P.hq.be, P.hq.w2t, P.hq.b2, qb);
    edge_mlp_kernel<HID><<<eblocks, 128, esm, stream>>>(
        xb, hb, edge_attr, ei, P.x2h_eww, P.x2h_ewb,
        P.hk.w1t, P.hk.b1, P.hk.g, P.hk.be, P.hk.w2t, P.hk.b2,
        P.hv.w1t, P.hv.b1, P.hv.g, P.hv.be, P.hv.w2t, P.hv.b2, kb, vb);
    attn_x2h_kernel<<<ablocks, 256, 0, stream>>>(qb, kb, vb, aggb);
    out_mlp_kernel<<<nblocks, 128, 0, stream>>>(aggb, hb, P.outm.w1t, P.outm.b1,
                                                P.outm.g, P.outm.be, P.outm.w2t,
                                                P.outm.b2);
    // ---- h2x ----
    node_mlp_kernel<<<nblocks, 128, 0, stream>>>(hb, P.xq.w1t, P.xq.b1, P.xq.g,
                                                 P.xq.be, P.xq.w2t, P.xq.b2, qb);
    edge_mlp_kernel<HEADS><<<eblocks, 128, esm, stream>>>(
        xb, hb, edge_attr, ei, P.h2x_eww, P.h2x_ewb,
        P.xk.w1t, P.xk.b1, P.xk.g, P.xk.be, P.xk.w2t, P.xk.b2,
        P.xv.w1t, P.xv.b1, P.xv.g, P.xv.be, P.xv.w2t, P.xv.b2, kb, vb);
    attn_h2x_kernel<<<ablocks, 256, 0, stream>>>(qb, kb, vb, xb, ei, dxh);
    x_update_kernel<<<(N_NODES + 255) / 256, 256, 0, stream>>>(xb, dxh);
  }

  final_copy_kernel<<<copy_blocks, 256, 0, stream>>>(hb, xb, (float*)d_out);
}
